// Grid2Mesh_26250840113767
// MI455X (gfx1250) — compile-verified, matrix operand order checked by reading, not run on hardware
//
#include <hip/hip_runtime.h>
#include <hip/hip_bf16.h>

// Problem constants (match reference)
#define NG 65536
#define NM 8192
#define E_EDGES 262144
#define KNBR 32
#define D 128

typedef __attribute__((ext_vector_type(16))) _Float16 v16h;
typedef __attribute__((ext_vector_type(8)))  float    v8f;

// ---------------------------------------------------------------------------
// Weight packing: row-major f32 (Ktot x 128) -> f16 WMMA B-fragment order.
// Fragment element index = ((kc*8 + nt)*32 + lane)*16 + j
//   col = nt*16 + (lane & 15)
//   kb  = (lane & 16) ? 8 : 0
//   k   = kc*32 + kb + (j < 8 ? j : j + 8)
// => every lane's fragment is 32 contiguous bytes.
// ---------------------------------------------------------------------------
__global__ __launch_bounds__(256) void pack_w_kernel(const float* __restrict__ W,
                                                     _Float16* __restrict__ out,
                                                     int total) {
  int idx = blockIdx.x * 256 + threadIdx.x;
  if (idx >= total) return;
  int j    = idx & 15;
  int lane = (idx >> 4) & 31;
  int nt   = (idx >> 9) & 7;
  int kc   = idx >> 12;
  int col  = nt * 16 + (lane & 15);
  int kb   = (lane & 16) ? 8 : 0;
  int k    = kc * 32 + kb + (j < 8 ? j : j + 8);
  out[idx] = (_Float16)W[k * D + col];
}

// ---------------------------------------------------------------------------
// 16xKtot (LDS, f16) @ Ktot x 128 (frag-packed, global) -> tanh -> C_lds f32
// 256 threads = 8 waves; wave nt owns output columns [nt*16, nt*16+16).
// ---------------------------------------------------------------------------
template <int KC>
__device__ __forceinline__ void mlp_tile_compute(const _Float16* A_lds, int ASTR,
                                                 float* C_lds, int CSTR,
                                                 const _Float16* __restrict__ wfrag) {
  const int tid  = threadIdx.x;
  const int lane = tid & 31;
  const int nt   = tid >> 5;  // wave id = N-tile

  v8f c = {};
#pragma unroll
  for (int kc = 0; kc < KC; ++kc) {
    // B fragment: 32 contiguous bytes per lane (coalesced, L2-hot)
    union { v16h h; uint4 u[2]; } b;
    const _Float16* bp = wfrag + (((kc * 8 + nt) * 32 + lane) << 4);
    b.u[0] = ((const uint4*)bp)[0];
    b.u[1] = ((const uint4*)bp)[1];
    // A fragment: two 16B LDS reads per ISA 16-bit A layout
    union { v16h h; uint4 u[2]; } a;
    const _Float16* ap = A_lds + (lane & 15) * ASTR + kc * 32 + ((lane & 16) ? 8 : 0);
    a.u[0] = *(const uint4*)ap;
    a.u[1] = *(const uint4*)(ap + 16);
    c = __builtin_amdgcn_wmma_f32_16x16x32_f16(false, a.h, false, b.h,
                                               (short)0, c, false, false);
  }
  // C layout: VGPR v, lanes 0-15 -> row v, lanes 16-31 -> row v+8; col = lane&15
  const int ccol = nt * 16 + (lane & 15);
  const int r0   = (lane >> 4) * 8;
#pragma unroll
  for (int v = 0; v < 8; ++v)
    C_lds[(r0 + v) * CSTR + ccol] = tanhf(c[v]);
}

// ---------------------------------------------------------------------------
// LayerNorm over 128 cols + gamma/beta + residual, straight to global.
// 16 threads per row; cross-thread reduce via shfl_xor within 16-lane groups.
// ---------------------------------------------------------------------------
__device__ __forceinline__ void ln_residual_store(const float* C_lds, int CSTR,
                                                  const float* __restrict__ g,
                                                  const float* __restrict__ bb,
                                                  const float* __restrict__ resid,
                                                  float* __restrict__ out) {
  const int tid = threadIdx.x;
  const int r   = tid >> 4;
  const int cg  = (tid & 15) * 8;
  float x[8];
  float s = 0.f, s2 = 0.f;
#pragma unroll
  for (int j = 0; j < 8; ++j) {
    x[j] = C_lds[r * CSTR + cg + j];
    s  += x[j];
    s2 += x[j] * x[j];
  }
#pragma unroll
  for (int off = 8; off >= 1; off >>= 1) {
    s  += __shfl_xor(s,  off, 16);
    s2 += __shfl_xor(s2, off, 16);
  }
  const float mu  = s * (1.f / 128.f);
  const float var = s2 * (1.f / 128.f) - mu * mu;
  const float rs  = rsqrtf(var + 1e-5f);
#pragma unroll
  for (int j = 0; j < 8; ++j) {
    const int col = cg + j;
    out[r * D + col] = (x[j] - mu) * rs * g[col] + bb[col] + resid[r * D + col];
  }
}

// ---------------------------------------------------------------------------
// Kernel 1: edge MLP.  A row = [bond | rect[p0] | mesh[p1]]  (Ktot = 384)
// out0 = bond + LN(tanh(A @ W_gm2e)) * g + b
// ---------------------------------------------------------------------------
#define ASTR1 (384 + 8)
__global__ __launch_bounds__(256) void edge_mlp_kernel(
    const float* __restrict__ bond, const float* __restrict__ rect,
    const float* __restrict__ mesh, const int* __restrict__ pair,
    const _Float16* __restrict__ wfrag, const float* __restrict__ g,
    const float* __restrict__ bb, float* __restrict__ out0) {
  __shared__ _Float16 A[16 * ASTR1];
  __shared__ float    C[16 * 132];
  __shared__ int      p0s[16], p1s[16];
  const int  tid = threadIdx.x;
  const long e0  = (long)blockIdx.x * 16;

  if (tid < 16) {
    p0s[tid] = pair[(e0 + tid) * 2 + 0];
    p1s[tid] = pair[(e0 + tid) * 2 + 1];
  }
  __syncthreads();

  // Stage A: 16 rows x 32 float4 per segment, 3 segments per index.
  for (int i = tid; i < 512; i += 256) {
    const int r  = i >> 5;
    const int c4 = (i & 31) * 4;
    const float4 f0 = *(const float4*)(bond + (e0 + r) * D + c4);
    const float4 f1 = *(const float4*)(rect + (long)p0s[r] * D + c4);
    const float4 f2 = *(const float4*)(mesh + (long)p1s[r] * D + c4);
    _Float16* arow = A + r * ASTR1;
    arow[c4 + 0] = (_Float16)f0.x; arow[c4 + 1] = (_Float16)f0.y;
    arow[c4 + 2] = (_Float16)f0.z; arow[c4 + 3] = (_Float16)f0.w;
    arow[128 + c4 + 0] = (_Float16)f1.x; arow[128 + c4 + 1] = (_Float16)f1.y;
    arow[128 + c4 + 2] = (_Float16)f1.z; arow[128 + c4 + 3] = (_Float16)f1.w;
    arow[256 + c4 + 0] = (_Float16)f2.x; arow[256 + c4 + 1] = (_Float16)f2.y;
    arow[256 + c4 + 2] = (_Float16)f2.z; arow[256 + c4 + 3] = (_Float16)f2.w;
  }
  __syncthreads();

  mlp_tile_compute<12>(A, ASTR1, C, 132, wfrag);
  __syncthreads();

  ln_residual_store(C, 132, g, bb, bond + e0 * D, out0 + e0 * D);
}

// ---------------------------------------------------------------------------
// Kernel 2: node aggregation + MLP. delta_edge recomputed as out0 - bond.
// A row = [mesh[n] | mean_k coef*delta_edge[eid]]  (Ktot = 256)
// ---------------------------------------------------------------------------
#define ASTR2 (256 + 8)
__global__ __launch_bounds__(256) void node_mlp_kernel(
    const float* __restrict__ mesh, const float* __restrict__ bond,
    const float* __restrict__ out0, const int* __restrict__ eid,
    const float* __restrict__ coef, const _Float16* __restrict__ wfrag,
    const float* __restrict__ g, const float* __restrict__ bb,
    float* __restrict__ out2) {
  __shared__ _Float16 A[16 * ASTR2];
  __shared__ float    C[16 * 132];
  __shared__ int      eids[16 * KNBR];
  __shared__ float    cfs[16 * KNBR];
  const int  tid = threadIdx.x;
  const long n0  = (long)blockIdx.x * 16;

  for (int i = tid; i < 16 * KNBR; i += 256) {
    eids[i] = eid[n0 * KNBR + i];
    cfs[i]  = coef[n0 * KNBR + i];
  }
  __syncthreads();

  const int r  = tid >> 4;
  const int cg = (tid & 15) * 8;
  float acc[8] = {};
  for (int k = 0; k < KNBR; ++k) {
    const long  e  = eids[r * KNBR + k];
    const float cf = cfs[r * KNBR + k];
    const float* pa = out0 + e * D + cg;
    const float* pb = bond + e * D + cg;
    const float4 a0 = *(const float4*)(pa + 0), a1 = *(const float4*)(pa + 4);
    const float4 b0 = *(const float4*)(pb + 0), b1 = *(const float4*)(pb + 4);
    acc[0] += cf * (a0.x - b0.x); acc[1] += cf * (a0.y - b0.y);
    acc[2] += cf * (a0.z - b0.z); acc[3] += cf * (a0.w - b0.w);
    acc[4] += cf * (a1.x - b1.x); acc[5] += cf * (a1.y - b1.y);
    acc[6] += cf * (a1.z - b1.z); acc[7] += cf * (a1.w - b1.w);
  }
  const float* mrow = mesh + (n0 + r) * D + cg;
  _Float16* arow = A + r * ASTR2;
#pragma unroll
  for (int j = 0; j < 8; ++j) {
    arow[cg + j]       = (_Float16)mrow[j];
    arow[128 + cg + j] = (_Float16)(acc[j] * (1.f / (float)KNBR));
  }
  __syncthreads();

  mlp_tile_compute<8>(A, ASTR2, C, 132, wfrag);
  __syncthreads();

  ln_residual_store(C, 132, g, bb, mesh + n0 * D, out2 + n0 * D);
}

// ---------------------------------------------------------------------------
// Kernel 3: rect MLP. A row = rect row (Ktot = 128)
// ---------------------------------------------------------------------------
#define ASTR3 (128 + 8)
__global__ __launch_bounds__(256) void rect_mlp_kernel(
    const float* __restrict__ rect, const _Float16* __restrict__ wfrag,
    const float* __restrict__ g, const float* __restrict__ bb,
    float* __restrict__ out1) {
  __shared__ _Float16 A[16 * ASTR3];
  __shared__ float    C[16 * 132];
  const int  tid = threadIdx.x;
  const long r0  = (long)blockIdx.x * 16;

  for (int i = tid; i < 512; i += 256) {
    const int r  = i >> 5;
    const int c4 = (i & 31) * 4;
    const float4 f = *(const float4*)(rect + (r0 + r) * D + c4);
    _Float16* arow = A + r * ASTR3 + c4;
    arow[0] = (_Float16)f.x; arow[1] = (_Float16)f.y;
    arow[2] = (_Float16)f.z; arow[3] = (_Float16)f.w;
  }
  __syncthreads();

  mlp_tile_compute<4>(A, ASTR3, C, 132, wfrag);
  __syncthreads();

  ln_residual_store(C, 132, g, bb, rect + r0 * D, out1 + r0 * D);
}

// ---------------------------------------------------------------------------
extern "C" void kernel_launch(void* const* d_in, const int* in_sizes, int n_in,
                              void* d_out, int out_size, void* d_ws, size_t ws_size,
                              hipStream_t stream) {
  const float* bond = (const float*)d_in[0];   // (E, 128)
  const float* rect = (const float*)d_in[1];   // (NG, 128)
  const float* mesh = (const float*)d_in[2];   // (NM, 128)
  const int*   pair = (const int*)d_in[3];     // (E, 2)
  const int*   eid  = (const int*)d_in[4];     // (NM, 32)
  const float* coef = (const float*)d_in[5];   // (NM, 32)
  const float* W1 = (const float*)d_in[6];     // (384,128)
  const float* g1 = (const float*)d_in[7];
  const float* b1 = (const float*)d_in[8];
  const float* W2 = (const float*)d_in[9];     // (256,128)
  const float* g2 = (const float*)d_in[10];
  const float* b2 = (const float*)d_in[11];
  const float* W3 = (const float*)d_in[12];    // (128,128)
  const float* g3 = (const float*)d_in[13];
  const float* b3 = (const float*)d_in[14];

  float* out0 = (float*)d_out;                        // E  x 128
  float* out1 = out0 + (size_t)E_EDGES * D;           // NG x 128
  float* out2 = out1 + (size_t)NG * D;                // NM x 128

  _Float16* wf1 = (_Float16*)d_ws;                    // 384*128 halves
  _Float16* wf2 = wf1 + 384 * D;                      // 256*128 halves
  _Float16* wf3 = wf2 + 256 * D;                      // 128*128 halves

  // Pack weights into WMMA fragment order (every launch; stream-ordered).
  pack_w_kernel<<<(384 * D) / 256, 256, 0, stream>>>(W1, wf1, 384 * D);
  pack_w_kernel<<<(256 * D) / 256, 256, 0, stream>>>(W2, wf2, 256 * D);
  pack_w_kernel<<<(128 * D) / 256, 256, 0, stream>>>(W3, wf3, 128 * D);

  // Edge MLP must precede node aggregation (same stream => ordered).
  edge_mlp_kernel<<<E_EDGES / 16, 256, 0, stream>>>(bond, rect, mesh, pair,
                                                    wf1, g1, b1, out0);
  rect_mlp_kernel<<<NG / 16, 256, 0, stream>>>(rect, wf3, g3, b3, out1);
  node_mlp_kernel<<<NM / 16, 256, 0, stream>>>(mesh, bond, out0, eid, coef,
                                               wf2, g2, b2, out2);
}
